// ODEGRUEncoder_28716151341121
// MI455X (gfx1250) — matrix operand order repaired and checked by reading, not compiled, not run on hardware
//
#include <hip/hip_runtime.h>
#include <hip/hip_fp16.h>
#include <math.h>

typedef _Float16 v8h  __attribute__((ext_vector_type(8)));
typedef _Float16 v16h __attribute__((ext_vector_type(16)));
typedef float    v8f  __attribute__((ext_vector_type(8)));

#define TILE_M 64
#define TILE_N 128
#define TILE_K 32
#define LDSB   40   // half pitch: 80B rows keep 16B alignment, stagger banks

// ---------------------------------------------------------------------------
// WMMA GEMM:  C[M,N] = act( A[M,K](f32) * W[N,K](f16)^T + bias[N] )
//   Double-buffered LDS pipeline:
//     - B tile (weights, pre-converted f16): global_load_async_to_lds_b128
//       (ASYNCcnt path, no VGPR round trip), waited with s_wait_asynccnt.
//     - A tile (activations f32): 2x global_load_b128 -> cvt -> ds_store_b128
//       per thread, issued for tile i+1 before computing tile i.
//   8 wave32 waves in a 2(M) x 4(N) grid, each computing 32x32 via
//   4x v_wmma_f32_16x16x32_f16 per K-chunk.
// ---------------------------------------------------------------------------
__global__ __launch_bounds__(256)
void gemm_wmma_f16(const float* __restrict__ A, int lda,
                   const _Float16* __restrict__ W,
                   const float* __restrict__ bias,
                   float* __restrict__ C, int ldc,
                   int M, int N, int K, int act)
{
    __shared__ __attribute__((aligned(16))) _Float16 As[2][TILE_M * LDSB];
    __shared__ __attribute__((aligned(16))) _Float16 Bs[2][TILE_N * LDSB];

    const int tid     = threadIdx.x;
    const int lane    = tid & 31;
    const int wave    = tid >> 5;
    const int wm      = wave & 1;      // wave M index (0..1)
    const int wn      = wave >> 1;     // wave N index (0..3)
    const int l16     = lane & 15;
    const int hsel    = lane >> 4;     // 0: lanes 0-15, 1: lanes 16-31
    const int blockM  = blockIdx.x * TILE_M;
    const int blockN  = blockIdx.y * TILE_N;

    // A staging: thread owns rows ar, k-chunk [ac, ac+8)
    const int ar = tid >> 2;           // 0..63
    const int ac = (tid & 3) * 8;      // 0,8,16,24
    const float* aptr = A + (size_t)(blockM + ar) * (size_t)lda + ac;

    v8f acc[2][2];
    {
        v8f z = {};
        #pragma unroll
        for (int i = 0; i < 2; ++i)
            #pragma unroll
            for (int j = 0; j < 2; ++j) acc[i][j] = z;
    }

    float4 areg0, areg1;

    auto loadA = [&](int kk) {
        const float4* p = (const float4*)(aptr + kk);
        areg0 = p[0];
        areg1 = p[1];
    };
    auto storeA = [&](int buf) {
        v8h h;
        h[0] = (_Float16)areg0.x; h[1] = (_Float16)areg0.y;
        h[2] = (_Float16)areg0.z; h[3] = (_Float16)areg0.w;
        h[4] = (_Float16)areg1.x; h[5] = (_Float16)areg1.y;
        h[6] = (_Float16)areg1.z; h[7] = (_Float16)areg1.w;
        *(v8h*)&As[buf][ar * LDSB + ac] = h;
    };
    // B tile: 128 rows x 4 chunks of 8 halfs = 512 slots, 2 per thread.
    // Async DMA straight into LDS (16B per lane per instruction).
    auto issueB = [&](int buf, int kk) {
        #pragma unroll
        for (int j = 0; j < 2; ++j) {
            int slot = tid + 256 * j;
            int row  = slot >> 2;
            int ch   = (slot & 3) * 8;
            const _Float16* g = W + (size_t)(blockN + row) * (size_t)K + kk + ch;
            unsigned loff = (unsigned)(size_t)&Bs[buf][row * LDSB + ch];
            asm volatile("global_load_async_to_lds_b128 %0, %1, off"
                         :: "v"(loff), "v"(g) : "memory");
        }
    };
    auto compute = [&](int buf) {
        #pragma unroll
        for (int mi = 0; mi < 2; ++mi) {
            // ISA 16-bit A 16x32 layout: lane group g holds K=g*8+0..7 in
            // halfs 0..7 and K=16+g*8+0..7 in halfs 8..15; row M = lane%16.
            const _Float16* ab = &As[buf][(wm * 32 + mi * 16 + l16) * LDSB + hsel * 8];
            v8h alo = *(const v8h*)ab;
            v8h ahi = *(const v8h*)(ab + 16);
            v16h a = __builtin_shufflevector(alo, ahi,
                0, 1, 2, 3, 4, 5, 6, 7, 8, 9, 10, 11, 12, 13, 14, 15);
            #pragma unroll
            for (int ni = 0; ni < 2; ++ni) {
                // B 32x16 layout matching A: lane group g holds K=g*8+0..7 in
                // halfs 0..7 and K=16+g*8+0..7 in halfs 8..15, column N = lane%16.
                const _Float16* bb = &Bs[buf][(wn * 32 + ni * 16 + l16) * LDSB + hsel * 8];
                v8h blo = *(const v8h*)bb;
                v8h bhi = *(const v8h*)(bb + 16);
                v16h b = __builtin_shufflevector(blo, bhi,
                    0, 1, 2, 3, 4, 5, 6, 7, 8, 9, 10, 11, 12, 13, 14, 15);
                acc[mi][ni] = __builtin_amdgcn_wmma_f32_16x16x32_f16(
                    false, a, false, b, (short)0, acc[mi][ni], false, false);
            }
        }
    };

    // ---- software pipeline --------------------------------------------
    loadA(0);
    issueB(0, 0);
    storeA(0);
    asm volatile("s_wait_asynccnt 0x0" ::: "memory");
    __syncthreads();

    const int nk = K / TILE_K;
    for (int ki = 0; ki < nk; ++ki) {
        const int cur = ki & 1;
        const int nxt = cur ^ 1;
        const bool more = (ki + 1) < nk;
        if (more) {
            loadA((ki + 1) * TILE_K);     // global loads overlap compute
            issueB(nxt, (ki + 1) * TILE_K); // async DMA overlaps compute
        }
        compute(cur);
        if (more) {
            storeA(nxt);
            asm volatile("s_wait_asynccnt 0x0" ::: "memory");
        }
        __syncthreads();
    }

    // f32 C/D layout: VGPR r -> M = r (lanes 0-15) / r+8 (lanes 16-31),
    // N = lane%16.
    #pragma unroll
    for (int mi = 0; mi < 2; ++mi) {
        #pragma unroll
        for (int ni = 0; ni < 2; ++ni) {
            const int col  = blockN + wn * 32 + ni * 16 + l16;
            const float bv = bias ? bias[col] : 0.0f;
            #pragma unroll
            for (int r = 0; r < 8; ++r) {
                const int row = blockM + wm * 32 + mi * 16 + r + hsel * 8;
                float v = acc[mi][ni][r] + bv;
                if (act == 1) v = tanhf(v);
                C[(size_t)row * (size_t)ldc + col] = v;
            }
        }
    }
}

// ---------------------------------------------------------------------------
__global__ __launch_bounds__(256)
void cvt_f32_to_f16(const float* __restrict__ src, _Float16* __restrict__ dst, int n)
{
    int i = (blockIdx.x * 256 + threadIdx.x) * 8;
    if (i >= n) return;
    const float4 a = *(const float4*)(src + i);
    const float4 b = *(const float4*)(src + i + 4);
    v8h h;
    h[0] = (_Float16)a.x; h[1] = (_Float16)a.y;
    h[2] = (_Float16)a.z; h[3] = (_Float16)a.w;
    h[4] = (_Float16)b.x; h[5] = (_Float16)b.y;
    h[6] = (_Float16)b.z; h[7] = (_Float16)b.w;
    *(v8h*)(dst + i) = h;
}

// dopri5 stage combine: y[i] = h[i] + hs * sum_j c_j * k_j[i]
// hs from device-side t[] (graph-capture safe).
__global__ __launch_bounds__(256)
void dp_combine(const float* h,
                const float* __restrict__ k0, const float* __restrict__ k1,
                const float* __restrict__ k2, const float* __restrict__ k3,
                const float* __restrict__ k4, const float* __restrict__ k5,
                float c0, float c1, float c2, float c3, float c4, float c5,
                int nterms, const float* __restrict__ tvec, int tIdx,
                float invSub, float* y, int n)
{
    int i = blockIdx.x * 256 + threadIdx.x;
    if (i >= n) return;
    const float hs = (tvec[tIdx] - tvec[tIdx - 1]) * invSub;
    float s = c0 * k0[i];
    if (nterms > 1) s += c1 * k1[i];
    if (nterms > 2) s += c2 * k2[i];
    if (nterms > 3) s += c3 * k3[i];
    if (nterms > 4) s += c4 * k4[i];
    if (nterms > 5) s += c5 * k5[i];
    y[i] = h[i] + hs * s;
}

__global__ __launch_bounds__(256)
void gru_gate(const float* __restrict__ gi, const float* __restrict__ gh,
              float* h, int Bn, int Hn)
{
    int i = blockIdx.x * 256 + threadIdx.x;
    if (i >= Bn * Hn) return;
    int b = i / Hn, j = i - b * Hn;
    const float* gib = gi + (size_t)b * 3 * Hn;
    const float* ghb = gh + (size_t)b * 3 * Hn;
    float ir = gib[j], iz = gib[Hn + j], in = gib[2 * Hn + j];
    float hr = ghb[j], hz = ghb[Hn + j], hn = ghb[2 * Hn + j];
    float r = 1.0f / (1.0f + expf(-(ir + hr)));
    float z = 1.0f / (1.0f + expf(-(iz + hz)));
    float nn = tanhf(in + r * hn);
    h[i] = (1.0f - z) * nn + z * h[i];
}

__global__ __launch_bounds__(256)
void fill_zero(float* p, int n)
{
    int i = blockIdx.x * 256 + threadIdx.x;
    if (i < n) p[i] = 0.0f;
}

// ---------------------------------------------------------------------------
static inline void launch_gemm(const float* A, int lda, const _Float16* W,
                               const float* bias, float* C, int ldc,
                               int M, int N, int K, int act, hipStream_t s)
{
    dim3 grid(M / TILE_M, N / TILE_N);
    gemm_wmma_f16<<<grid, dim3(256), 0, s>>>(A, lda, W, bias, C, ldc, M, N, K, act);
}

extern "C" void kernel_launch(void* const* d_in, const int* in_sizes, int n_in,
                              void* d_out, int out_size, void* d_ws, size_t ws_size,
                              hipStream_t stream)
{
    const int B = 512, S = 128, D = 128, H = 1024, L = 128;
    const float* x    = (const float*)d_in[0];
    const float* tv   = (const float*)d_in[1];
    const float* w_ih = (const float*)d_in[2];
    const float* w_hh = (const float*)d_in[3];
    const float* b_ih = (const float*)d_in[4];
    const float* b_hh = (const float*)d_in[5];
    const float* w0   = (const float*)d_in[6];
    const float* b0   = (const float*)d_in[7];
    const float* w1   = (const float*)d_in[8];
    const float* b1   = (const float*)d_in[9];
    const float* w2   = (const float*)d_in[10];
    const float* b2   = (const float*)d_in[11];
    const float* mu_w = (const float*)d_in[12];
    const float* mu_b = (const float*)d_in[13];
    const float* lv_w = (const float*)d_in[14];
    const float* lv_b = (const float*)d_in[15];
    float* out = (float*)d_out;

    // fp32 scratch (~33.5 MB) -----------------------------------------------
    float* ws = (float*)d_ws;
    size_t off = 0;
    float* h  = ws + off; off += (size_t)B * H;
    float* y  = ws + off; off += (size_t)B * H;
    float* t1 = ws + off; off += (size_t)B * H;
    float* t2 = ws + off; off += (size_t)B * H;
    float* kb[6];
    for (int i = 0; i < 6; ++i) { kb[i] = ws + off; off += (size_t)B * H; }
    float* gi = ws + off; off += (size_t)B * 3 * H;
    float* gh = ws + off; off += (size_t)B * 3 * H;

    // fp16 weight copies (~14 MB), converted once per call ------------------
    _Float16* hbase = (_Float16*)(ws + off);
    size_t hoff = 0;
    _Float16* w_ih_h = hbase + hoff; hoff += (size_t)3 * H * D;
    _Float16* w_hh_h = hbase + hoff; hoff += (size_t)3 * H * H;
    _Float16* w0_h   = hbase + hoff; hoff += (size_t)H * H;
    _Float16* w1_h   = hbase + hoff; hoff += (size_t)H * H;
    _Float16* w2_h   = hbase + hoff; hoff += (size_t)H * H;
    _Float16* mu_w_h = hbase + hoff; hoff += (size_t)L * H;
    _Float16* lv_w_h = hbase + hoff; hoff += (size_t)L * H;

    auto cvt = [&](const float* s, _Float16* d, int n) {
        cvt_f32_to_f16<<<(n / 8 + 255) / 256, 256, 0, stream>>>(s, d, n);
    };
    cvt(w_ih, w_ih_h, 3 * H * D);
    cvt(w_hh, w_hh_h, 3 * H * H);
    cvt(w0,   w0_h,   H * H);
    cvt(w1,   w1_h,   H * H);
    cvt(w2,   w2_h,   H * H);
    cvt(mu_w, mu_w_h, L * H);
    cvt(lv_w, lv_w_h, L * H);

    const int n  = B * H;
    const int eB = (n + 255) / 256;

    // Dormand–Prince rows 2..6 plus b-weights. FSAL: h_new == stage-7 input
    // combine (row == DP_B[:6]), so k6 is never computed.
    static const float rows[6][6] = {
        {1.0f/5,          0,               0,               0,            0,                0},
        {3.0f/40,         9.0f/40,         0,               0,            0,                0},
        {44.0f/45,       -56.0f/15,        32.0f/9,         0,            0,                0},
        {19372.0f/6561,  -25360.0f/2187,   64448.0f/6561,  -212.0f/729,   0,                0},
        {9017.0f/3168,   -355.0f/33,       46732.0f/5247,   49.0f/176,   -5103.0f/18656,    0},
        {35.0f/384,       0.0f,            500.0f/1113,     125.0f/192,  -2187.0f/6784,     11.0f/84},
    };

    auto mlp = [&](const float* in, float* kout) {
        launch_gemm(in, H, w0_h, b0, t1,   H, B, H, H, 1, stream);
        launch_gemm(t1, H, w1_h, b1, t2,   H, B, H, H, 1, stream);
        launch_gemm(t2, H, w2_h, b2, kout, H, B, H, H, 0, stream);
    };

    // h = GRU(x[:,0,:], 0)
    fill_zero<<<eB, 256, 0, stream>>>(h, n);
    launch_gemm(x, S * D, w_ih_h, b_ih, gi, 3 * H, B, 3 * H, D, 0, stream);
    launch_gemm(h, H,     w_hh_h, b_hh, gh, 3 * H, B, 3 * H, H, 0, stream);
    gru_gate<<<eB, 256, 0, stream>>>(gi, gh, h, B, H);

    for (int ti = 1; ti < S; ++ti) {
        for (int sub = 0; sub < 4; ++sub) {
            mlp(h, kb[0]);                                   // k0 = f(h)
            for (int st = 1; st <= 5; ++st) {
                const float* r = rows[st - 1];
                dp_combine<<<eB, 256, 0, stream>>>(
                    h, kb[0], kb[1], kb[2], kb[3], kb[4], kb[5],
                    r[0], r[1], r[2], r[3], r[4], r[5], st,
                    tv, ti, 0.25f, y, n);
                mlp(y, kb[st]);                              // k_st = f(y_st)
            }
            const float* r = rows[5];                        // b-weights (FSAL)
            dp_combine<<<eB, 256, 0, stream>>>(
                h, kb[0], kb[1], kb[2], kb[3], kb[4], kb[5],
                r[0], r[1], r[2], r[3], r[4], r[5], 6,
                tv, ti, 0.25f, h, n);                        // in-place update
        }
        launch_gemm(x + (size_t)ti * D, S * D, w_ih_h, b_ih, gi, 3 * H, B, 3 * H, D, 0, stream);
        launch_gemm(h, H,               w_hh_h, b_hh, gh, 3 * H, B, 3 * H, H, 0, stream);
        gru_gate<<<eB, 256, 0, stream>>>(gi, gh, h, B, H);
    }

    launch_gemm(h, H, mu_w_h, mu_b, out,                 L, B, L, H, 0, stream);
    launch_gemm(h, H, lv_w_h, lv_b, out + (size_t)B * L, L, B, L, H, 0, stream);
}
